// BPNN_75359496175846
// MI455X (gfx1250) — matrix operand order repaired and checked by reading, not compiled, not run on hardware
//
#include <hip/hip_runtime.h>
#include <hip/hip_bf16.h>

typedef __attribute__((ext_vector_type(16))) _Float16 v16h;
typedef __attribute__((ext_vector_type(8)))  float    v8f;

#define N_ATOMS_C       131072
#define IN_DIM          128
#define HID             256
#define N_ELEM_C        4
#define BLOCK_THREADS   256
#define WAVES           8
#define ATOMS_PER_BLOCK 1024
#define TILES_PER_WAVE  (ATOMS_PER_BLOCK / (WAVES * 16))   // 8

// ---- dynamic LDS layout (bytes) ----
#define SZ_W0   (16 * 4 * 32 * 16 * 2)   // 65536 : W0 f16, fragment-major [nt][kb][lane][i]
#define SZ_W1   (16 * 8 * 32 * 16 * 2)   // 131072: W1 f16, fragment-major
#define SZ_B0   (HID * 4)                // 1024
#define SZ_B1   (HID * 4)                // 1024
#define SZ_W2   (HID * 4)                // 1024
#define SZ_FST  (16 * IN_DIM * 2)        // 4096 per wave
#define SZ_HST  (16 * HID * 2)           // 8192 per wave
#define OFF_W0  0
#define OFF_W1  (OFF_W0 + SZ_W0)
#define OFF_B0  (OFF_W1 + SZ_W1)
#define OFF_B1  (OFF_B0 + SZ_B0)
#define OFF_W2  (OFF_B1 + SZ_B1)
#define OFF_FST (OFF_W2 + SZ_W2)
#define OFF_HST (OFF_FST + WAVES * SZ_FST)
#define SMEM_TOTAL (OFF_HST + WAVES * SZ_HST)              // 297984 B < 320 KB

static __device__ inline unsigned pack2h(float a, float b) {
    _Float16 ha = (_Float16)a, hb = (_Float16)b;
    unsigned short ua = __builtin_bit_cast(unsigned short, ha);
    unsigned short ub = __builtin_bit_cast(unsigned short, hb);
    return (unsigned)ua | ((unsigned)ub << 16);
}

// Branch-free tanh on hardware transcendentals:
// tanh(x) = 1 - 2 / (exp(2x) + 1); exp(2x) = exp2(x * 2*log2(e)).
// Saturates correctly: x>>0 -> exp2=inf -> 1; x<<0 -> exp2=0 -> -1.
static __device__ inline float fast_tanh(float x) {
    float t = __builtin_amdgcn_exp2f(x * 2.885390081777927f);
    return 1.0f - 2.0f * __builtin_amdgcn_rcpf(t + 1.0f);
}

// A-fragment (16x32 f16): lane holds row M=lane&15; kg=lane>>4.
// halves i=0..15 -> K = kb*32 + kg*16 + i
// -> one contiguous 32-byte run per lane.
static __device__ inline v16h load_a_frag(const _Float16* base, int row, int kg,
                                          int kb, int rowlen_halves) {
    union { uint4 u[2]; v16h v; } t;
    const uint4* p = (const uint4*)base;
    int idx = row * (rowlen_halves >> 3) + kb * 4 + kg * 2;
    t.u[0] = p[idx];
    t.u[1] = p[idx + 1];
    return t.v;
}

__global__ __launch_bounds__(BLOCK_THREADS)
void bpnn_moe_wmma(const float* __restrict__ F,
                   const float* __restrict__ W0g, const float* __restrict__ b0g,
                   const float* __restrict__ W1g, const float* __restrict__ b1g,
                   const float* __restrict__ W2g, const float* __restrict__ b2g,
                   const float* __restrict__ maskw,
                   const int*  __restrict__ zat,  const int* __restrict__ imgidx,
                   float* __restrict__ out)
{
    extern __shared__ char smem[];
    _Float16* w0f    = (_Float16*)(smem + OFF_W0);
    _Float16* w1f    = (_Float16*)(smem + OFF_W1);
    float*    b0s    = (float*)(smem + OFF_B0);
    float*    b1s    = (float*)(smem + OFF_B1);
    float*    w2s    = (float*)(smem + OFF_W2);

    const int e    = blockIdx.y;
    const int tid  = threadIdx.x;
    const int lane = tid & 31;
    const int wave = tid >> 5;
    const int n16  = lane & 15;
    const int kg   = lane >> 4;

    _Float16* fstage = (_Float16*)(smem + OFF_FST) + wave * (16 * IN_DIM);
    _Float16* hstage = (_Float16*)(smem + OFF_HST) + wave * (16 * HID);

    // ---- stage expert weights into LDS, B-fragment-major, f32 -> f16 ----
    for (int id = tid; id < 16 * 4 * 32 * 16; id += BLOCK_THREADS) {
        int i  = id & 15;
        int ln = (id >> 4) & 31;
        int kb = (id >> 9) & 3;
        int nt = id >> 11;
        int k  = kb * 32 + (ln >> 4) * 16 + i;
        int n  = nt * 16 + (ln & 15);
        w0f[id] = (_Float16)W0g[((size_t)e * IN_DIM + k) * HID + n];
    }
    for (int id = tid; id < 16 * 8 * 32 * 16; id += BLOCK_THREADS) {
        int i  = id & 15;
        int ln = (id >> 4) & 31;
        int kb = (id >> 9) & 7;
        int nt = id >> 12;
        int k  = kb * 32 + (ln >> 4) * 16 + i;
        int n  = nt * 16 + (ln & 15);
        w1f[id] = (_Float16)W1g[((size_t)e * HID + k) * HID + n];
    }
    if (tid < HID) {
        b0s[tid] = b0g[e * HID + tid];
        b1s[tid] = b1g[e * HID + tid];
        w2s[tid] = W2g[e * HID + tid];
    }
    __syncthreads();

    const float bias2 = b2g[e];
    const v16h* w0v = (const v16h*)w0f;
    const v16h* w1v = (const v16h*)w1f;

    for (int t = 0; t < TILES_PER_WAVE; ++t) {
        const int atomBase = blockIdx.x * ATOMS_PER_BLOCK + wave * (TILES_PER_WAVE * 16) + t * 16;

        // ---- stage 16 fingerprint rows as f16 (lane: row=n16, col half kg) ----
        {
            const int row = n16;
            const float4* src = (const float4*)(F + (size_t)(atomBase + row) * IN_DIM + kg * 64);
            unsigned* dst = (unsigned*)fstage + row * (IN_DIM / 2) + kg * 32;
            #pragma unroll
            for (int j = 0; j < 16; ++j) {
                float4 f = src[j];
                dst[j * 2 + 0] = pack2h(f.x, f.y);
                dst[j * 2 + 1] = pack2h(f.z, f.w);
            }
        }

        // ---- layer 0: [16x128] @ [128x256] + b0, tanh -> hstage (f16) ----
        #pragma unroll 2
        for (int nt = 0; nt < 16; ++nt) {
            float bv = b0s[nt * 16 + n16];
            v8f acc;
            #pragma unroll
            for (int r = 0; r < 8; ++r) acc[r] = bv;
            #pragma unroll
            for (int kb = 0; kb < 4; ++kb) {
                v16h a = load_a_frag(fstage, n16, kg, kb, IN_DIM);
                v16h b = w0v[(nt * 4 + kb) * 32 + lane];
                acc = __builtin_amdgcn_wmma_f32_16x16x32_f16(
                        false, a, false, b, (short)0, acc, false, false);
            }
            // C layout: VGPR r -> row kg*8+r, col nt*16+n16
            #pragma unroll
            for (int r = 0; r < 8; ++r)
                hstage[(kg * 8 + r) * HID + nt * 16 + n16] = (_Float16)fast_tanh(acc[r]);
        }

        // ---- layer 1: [16x256] @ [256x256] + b1, tanh, fused dot with W2 ----
        float po[8];
        #pragma unroll
        for (int r = 0; r < 8; ++r) po[r] = 0.0f;

        #pragma unroll 2
        for (int nt = 0; nt < 16; ++nt) {
            float bv = b1s[nt * 16 + n16];
            v8f acc;
            #pragma unroll
            for (int r = 0; r < 8; ++r) acc[r] = bv;
            #pragma unroll
            for (int kb = 0; kb < 8; ++kb) {
                v16h a = load_a_frag(hstage, n16, kg, kb, HID);
                v16h b = w1v[(nt * 8 + kb) * 32 + lane];
                acc = __builtin_amdgcn_wmma_f32_16x16x32_f16(
                        false, a, false, b, (short)0, acc, false, false);
            }
            float wv = w2s[nt * 16 + n16];
            #pragma unroll
            for (int r = 0; r < 8; ++r)
                po[r] += fast_tanh(acc[r]) * wv;
        }

        // ---- reduce columns across each 16-lane half (wave32-safe) ----
        #pragma unroll
        for (int r = 0; r < 8; ++r) {
            po[r] += __shfl_xor(po[r], 1, 32);
            po[r] += __shfl_xor(po[r], 2, 32);
            po[r] += __shfl_xor(po[r], 4, 32);
            po[r] += __shfl_xor(po[r], 8, 32);
        }

        // ---- mask-select expert + scatter into per-image energy ----
        if (n16 == 0) {
            #pragma unroll
            for (int r = 0; r < 8; ++r) {
                int atom = atomBase + kg * 8 + r;
                int z = zat[atom];
                float m = maskw[z * N_ELEM_C + e];
                if (m != 0.0f)
                    atomicAdd(&out[imgidx[atom]], (po[r] + bias2) * m);
            }
        }
    }
}

extern "C" void kernel_launch(void* const* d_in, const int* in_sizes, int n_in,
                              void* d_out, int out_size, void* d_ws, size_t ws_size,
                              hipStream_t stream) {
    (void)in_sizes; (void)n_in; (void)d_ws; (void)ws_size;
    const float* F     = (const float*)d_in[0];
    const float* W0g   = (const float*)d_in[1];
    const float* b0g   = (const float*)d_in[2];
    const float* W1g   = (const float*)d_in[3];
    const float* b1g   = (const float*)d_in[4];
    const float* W2g   = (const float*)d_in[5];
    const float* b2g   = (const float*)d_in[6];
    const float* maskw = (const float*)d_in[7];
    const int*   zat   = (const int*)d_in[8];
    const int*   img   = (const int*)d_in[9];
    float* out = (float*)d_out;

    // d_out is poisoned by the harness; atomics need zeros (capture-safe async memset)
    hipMemsetAsync(out, 0, (size_t)out_size * sizeof(float), stream);

    dim3 grid(N_ATOMS_C / ATOMS_PER_BLOCK, N_ELEM_C);  // (128, 4)
    bpnn_moe_wmma<<<grid, BLOCK_THREADS, SMEM_TOTAL, stream>>>(
        F, W0g, b0g, W1g, b1g, W2g, b2g, maskw, zat, img, out);
}
